// Net_48962627175132
// MI455X (gfx1250) — matrix operand order repaired and checked by reading, not compiled, not run on hardware
//
#include <hip/hip_runtime.h>
#include <math.h>

// ---------------------------------------------------------------------------
// DimeNet-style forward for MI455X (gfx1250, wave32, WMMA).
//   * dense layers + bilinear einsum -> v_wmma_f32_16x16x32_f16
//   * weights pre-converted (batched) to LDS-ready transposed f16; B tiles
//     staged with global_load_async_to_lds_b128 (ASYNCcnt)
//   * double-buffered LDS: one barrier per K-step, stage(k+1) overlaps
//     compute(k); full-tile fast paths avoid per-element exec guards
//   * A tiles: float4 loads + v_cvt_pk_rtz_f16_f32 + b64 LDS stores
//   * fragments: ds_load_b128 pairs; 4 WMMAs per K-step per wave
//   * segment sums fused as global_atomic_add_f32
// ---------------------------------------------------------------------------

#define H 128
#define NB 8
#define NS 7
#define NR 6
#define CUTOFF 5.0f
#define LDSTRIDE 48

typedef __attribute__((ext_vector_type(16))) _Float16 v16h;
typedef __attribute__((ext_vector_type(8)))  _Float16 v8h;
typedef __attribute__((ext_vector_type(4)))  _Float16 v4h;
typedef __attribute__((ext_vector_type(2)))  _Float16 v2h;
typedef __attribute__((ext_vector_type(8)))  float    v8f;

union H16 { v16h v; v8h q[2]; _Float16 h[16]; };
union F8  { v8f  v; float f[8]; };
union V4P { v4h v4; v2h v2[2]; };

__device__ __forceinline__ v2h pk2(float a, float b) {
  auto r = __builtin_amdgcn_cvt_pkrtz(a, b);   // v_cvt_pk_rtz_f16_f32
  union { decltype(r) rv; v2h hv; } u; u.rv = r;
  return u.hv;
}

__device__ __forceinline__ float silu_f(float v) { return v / (1.f + __expf(-v)); }

// Async copy of 16 contiguous bytes global -> LDS (tracked by ASYNCcnt).
__device__ __forceinline__ void async_b128(void* lds_ptr, const void* gbase, unsigned goff) {
  unsigned lds = (unsigned)(uintptr_t)lds_ptr;   // low 32 bits of flat == LDS offset
  asm volatile("global_load_async_to_lds_b128 %0, %1, %2"
               :: "v"(lds), "v"(goff), "s"(gbase) : "memory");
}
__device__ __forceinline__ void wait_async0() {
  asm volatile("s_wait_asynccnt 0x0" ::: "memory");
}

// Spherical Bessel zeros z_{l,n} (l=0..6, n=1..6)
__constant__ float Zc[NS][NR] = {
  { 3.14159265f,  6.28318531f,  9.42477796f, 12.56637061f, 15.70796327f, 18.84955592f},
  { 4.49340946f,  7.72525184f, 10.90412166f, 14.06619391f, 17.22075527f, 20.37130296f},
  { 5.76345920f,  9.09501133f, 12.32294097f, 15.51460301f, 18.68903635f, 21.85387422f},
  { 6.98793200f, 10.41711855f, 13.69802315f, 16.92362129f, 20.12180617f, 23.30424707f},
  { 8.18256145f, 11.70490715f, 15.03966471f, 18.30125596f, 21.52541754f, 24.72756582f},
  { 9.35581211f, 12.96653017f, 16.35470964f, 19.65315286f, 22.90455064f, 26.12775036f},
  {10.51283540f, 14.20739214f, 17.64797496f, 20.98346327f, 24.26276804f, 27.50786836f}};

// sqrt((2l+1)/(4*pi))
__constant__ float Ccoef[NS] = {0.28209479f, 0.48860251f, 0.63078313f, 0.74635267f,
                                0.84628438f, 0.93560258f, 1.01782703f};

__device__ float sph_jl(int l, float x) {
  float s = __sinf(x), c = __cosf(x);
  float j0 = s / x;
  if (l == 0) return j0;
  float jm = j0, jc = s / (x * x) - c / x;
  for (int ll = 2; ll <= l; ++ll) {
    float t = (2.f * ll - 1.f) / x * jc - jm;
    jm = jc; jc = t;
  }
  return jc;
}

// ---------------------------------------------------------------------------
// Batched weight conversions: W[K,N] f32 -> Wt[N][K] f16 (K contiguous)
// ---------------------------------------------------------------------------
#define MAXW 80
struct ConvTab  { const float* src[MAXW]; _Float16* dst[MAXW]; };
struct ConvFTab { const float* src[6];    _Float16* dst[6]; };

__global__ void convT128_kernel(ConvTab tab) {   // all K=128, N=128 weights
  const float* W  = tab.src[blockIdx.y];
  _Float16*    Wt = tab.dst[blockIdx.y];
  int idx = blockIdx.x * blockDim.x + threadIdx.x;   // exactly 128*64 threads
  int n = idx >> 6, k2 = (idx & 63) << 1;
  *(v2h*)&Wt[(size_t)n * H + k2] = pk2(W[(size_t)k2 * H + n], W[(size_t)(k2 + 1) * H + n]);
}

__global__ void convT_kernel(const float* __restrict__ W, _Float16* __restrict__ Wt,
                             int K, int N) {      // generic (emb_lin, K=384)
  int idx = blockIdx.x * blockDim.x + threadIdx.x;
  int kh = K >> 1;
  if (idx >= N * kh) return;
  int n = idx / kh, k2 = (idx % kh) * 2;
  *(v2h*)&Wt[(size_t)n * K + k2] = pk2(W[(size_t)k2 * N + n], W[(size_t)(k2 + 1) * N + n]);
}

__global__ void convF_batch_kernel(ConvFTab tab) {  // W_bil flat f32->f16
  const float* W  = tab.src[blockIdx.y];
  _Float16*    Wt = tab.dst[blockIdx.y];
  int idx = blockIdx.x * blockDim.x + threadIdx.x;  // exactly (128*1024)/2 threads
  const float2 f = *(const float2*)&W[(size_t)idx * 2];
  *(v2h*)&Wt[(size_t)idx * 2] = pk2(f.x, f.y);
}

// ---------------------------------------------------------------------------
// Geometry / gather / scatter kernels
// ---------------------------------------------------------------------------
__global__ void norm_kernel(float* norm42) {
  int id = threadIdx.x;
  if (id >= NS * NR) return;
  int l = id / NR;
  float x = Zc[l][id % NR];
  norm42[id] = sqrtf(2.f) / fabsf(sph_jl(l + 1, x));
}

__global__ void edge_kernel(const float* __restrict__ pos,
                            const int* __restrict__ src, const int* __restrict__ dst,
                            const float* __restrict__ freq, const float* __restrict__ norm42,
                            float* __restrict__ rbf, float* __restrict__ rad, int E) {
  int e = blockIdx.x * blockDim.x + threadIdx.x;
  if (e >= E) return;
  int s = src[e], t = dst[e];
  float dx = pos[s * 3 + 0] - pos[t * 3 + 0];
  float dy = pos[s * 3 + 1] - pos[t * 3 + 1];
  float dz = pos[s * 3 + 2] - pos[t * 3 + 2];
  float d = sqrtf(dx * dx + dy * dy + dz * dz) * (1.f / CUTOFF);
  float env = 0.f;
  if (d < 1.f) {
    float d2 = d * d, d5 = d2 * d2 * d;
    env = 1.f / d - 28.f * d5 + 48.f * d5 * d - 21.f * d5 * d2;
  }
  for (int n = 0; n < NR; ++n) rbf[(size_t)e * NR + n] = env * __sinf(freq[n] * d);
  for (int l = 0; l < NS; ++l)
    for (int n = 0; n < NR; ++n)
      rad[(size_t)e * 42 + l * NR + n] = env * norm42[l * NR + n] * sph_jl(l, d * Zc[l][n]);
}

__global__ void trip_kernel(const float* __restrict__ pos,
                            const int* __restrict__ src, const int* __restrict__ dst,
                            const int* __restrict__ ikj, const int* __restrict__ iji,
                            float* __restrict__ cbf, int T) {
  int w = blockIdx.x * blockDim.x + threadIdx.x;
  if (w >= T) return;
  int eji = iji[w], ekj = ikj[w];
  int j = src[eji], i = dst[eji], k = src[ekj];
  float v1x = pos[i * 3 + 0] - pos[j * 3 + 0];
  float v1y = pos[i * 3 + 1] - pos[j * 3 + 1];
  float v1z = pos[i * 3 + 2] - pos[j * 3 + 2];
  float v2x = pos[k * 3 + 0] - pos[j * 3 + 0];
  float v2y = pos[k * 3 + 1] - pos[j * 3 + 1];
  float v2z = pos[k * 3 + 2] - pos[j * 3 + 2];
  float aa = v1x * v2x + v1y * v2y + v1z * v2z;
  float cx = v1y * v2z - v1z * v2y;
  float cy = v1z * v2x - v1x * v2z;
  float cz = v1x * v2y - v1y * v2x;
  float bb2 = cx * cx + cy * cy + cz * cz;
  float ct = aa * rsqrtf(aa * aa + bb2 + 1e-30f);
  float Pm = 1.f, Pc = ct;
  cbf[(size_t)w * NS + 0] = Ccoef[0];
  cbf[(size_t)w * NS + 1] = Ccoef[1] * ct;
  for (int l = 2; l < NS; ++l) {
    float Pn = ((2.f * l - 1.f) * ct * Pc - (l - 1.f) * Pm) / (float)l;
    Pm = Pc; Pc = Pn;
    cbf[(size_t)w * NS + l] = Ccoef[l] * Pc;
  }
}

__global__ void sbfp_kernel(const float* __restrict__ rad, const float* __restrict__ cbf,
                            const int* __restrict__ ikj, const float* __restrict__ Wsbf,
                            float* __restrict__ sbfp, int T) {
  int w = blockIdx.x * blockDim.x + threadIdx.x;
  if (w >= T) return;
  int kj = ikj[w];
  float acc[NB];
#pragma unroll
  for (int j = 0; j < NB; ++j) acc[j] = 0.f;
  for (int l = 0; l < NS; ++l) {
    float cb = cbf[(size_t)w * NS + l];
    for (int n = 0; n < NR; ++n) {
      float rv = rad[(size_t)kj * 42 + l * NR + n] * cb;
#pragma unroll
      for (int j = 0; j < NB; ++j) acc[j] += rv * Wsbf[(l * NR + n) * NB + j];
    }
  }
#pragma unroll
  for (int j = 0; j < NB; ++j) sbfp[(size_t)w * NB + j] = acc[j];
}

__global__ void concat_kernel(const float* __restrict__ emb, const int* __restrict__ z,
                              const int* __restrict__ src, const int* __restrict__ dst,
                              const float* __restrict__ rbfe, float* __restrict__ cat, int E) {
  long idx = blockIdx.x * (long)blockDim.x + threadIdx.x;
  if (idx >= (long)E * H) return;
  int e = (int)(idx >> 7), h = (int)(idx & 127);
  cat[(size_t)e * 384 + h]       = emb[(size_t)z[dst[e]] * H + h];
  cat[(size_t)e * 384 + 128 + h] = emb[(size_t)z[src[e]] * H + h];
  cat[(size_t)e * 384 + 256 + h] = rbfe[idx];
}

__global__ void segsum_kernel(const float* __restrict__ g, const int* __restrict__ dst,
                              float* __restrict__ out, int E) {
  long idx = blockIdx.x * (long)blockDim.x + threadIdx.x;
  if (idx >= (long)E * H) return;
  int e = (int)(idx >> 7);
  atomicAdd(&out[(size_t)dst[e] * H + (idx & 127)], g[idx]);
}

__global__ void final_kernel(const float* __restrict__ t, const float* __restrict__ Wf,
                             const int* __restrict__ batch, float* __restrict__ P, int n) {
  int r = blockIdx.x * blockDim.x + threadIdx.x;
  if (r >= n) return;
  float acc = 0.f;
#pragma unroll 4
  for (int h = 0; h < H; ++h) acc += t[(size_t)r * H + h] * Wf[h];
  atomicAdd(&P[batch[r]], acc);
}

// ---------------------------------------------------------------------------
// WMMA GEMM, f16 pre-transposed weights, async B staging, LDS double buffer.
// C[M,128] = epi(A[M,K] @ W); Wt is [128][K] f16, K multiple of 32.
// Block tile 128x64 (grid.y = 2), 8 waves, 4 accumulators/wave.
// Loop shape: wait_asynccnt 0 -> barrier -> stage(k+1) -> compute(k).
// FLAGS: 1=+bias  2=silu  4=*mulp  8=+addp
// ---------------------------------------------------------------------------
template <int FLAGS>
__global__ __launch_bounds__(256) void gemmh_kernel(
    const float* __restrict__ A, const _Float16* __restrict__ Wt,
    const float* __restrict__ bias, const float* __restrict__ mulp,
    const float* __restrict__ addp, float* __restrict__ C,
    int M, int K) {
  __shared__ _Float16 As[2][128][LDSTRIDE];
  __shared__ _Float16 Bs[2][64][LDSTRIDE];
  int tid = threadIdx.x, lane = tid & 31, wv = tid >> 5;
  int m0 = blockIdx.x * 128, n0 = blockIdx.y * 64;
  bool full = (m0 + 128 <= M);
  F8 acc[4];
#pragma unroll
  for (int t = 0; t < 4; ++t)
#pragma unroll
    for (int r = 0; r < 8; ++r) acc[t].f[r] = 0.f;

  int bn_row = tid >> 2, bn_ch = (tid & 3) << 3;  // async B: 64 rows x 4 b128 chunks
  int ar = tid >> 3, ac4 = (tid & 7) << 2;        // A staging coords (iter 0)

  auto stage = [&](int k0, int buf) {
    async_b128(&Bs[buf][bn_row][bn_ch], Wt,
               (unsigned)(((size_t)(n0 + bn_row) * K + k0 + bn_ch) * sizeof(_Float16)));
    if (full) {
#pragma unroll
      for (int it = 0; it < 4; ++it) {
        int r = ar + it * 32;
        const float4 f = *(const float4*)&A[(size_t)(m0 + r) * K + k0 + ac4];
        V4P u;
        u.v2[0] = pk2(f.x, f.y);
        u.v2[1] = pk2(f.z, f.w);
        *(v4h*)&As[buf][r][ac4] = u.v4;
      }
    } else {
#pragma unroll
      for (int it = 0; it < 4; ++it) {
        int r = ar + it * 32;
        int gr = m0 + r;
        V4P u;
        if (gr < M) {
          const float4 f = *(const float4*)&A[(size_t)gr * K + k0 + ac4];
          u.v2[0] = pk2(f.x, f.y);
          u.v2[1] = pk2(f.z, f.w);
        } else {
          u.v2[0] = pk2(0.f, 0.f);
          u.v2[1] = pk2(0.f, 0.f);
        }
        *(v4h*)&As[buf][r][ac4] = u.v4;
      }
    }
  };

  stage(0, 0);
  int cur = 0;
  for (int k0 = 0; k0 < K; k0 += 32, cur ^= 1) {
    wait_async0();
    __syncthreads();
    if (k0 + 32 < K) stage(k0 + 32, cur ^ 1);

    H16 a, b[4];
    int am = (wv << 4) + (lane & 15);
    int koffA = (lane & 16) ? 8 : 0;
    a.q[0] = *(const v8h*)&As[cur][am][koffA];
    a.q[1] = *(const v8h*)&As[cur][am][koffA + 16];
    int bn = lane & 15, koffB = (lane & 16) ? 8 : 0;
#pragma unroll
    for (int t = 0; t < 4; ++t) {
      b[t].q[0] = *(const v8h*)&Bs[cur][bn + 16 * t][koffB];
      b[t].q[1] = *(const v8h*)&Bs[cur][bn + 16 * t][koffB + 16];
    }
#pragma unroll
    for (int t = 0; t < 4; ++t)
      acc[t].v = __builtin_amdgcn_wmma_f32_16x16x32_f16(false, a.v, false, b[t].v,
                                                        (short)0, acc[t].v, false, false);
  }

  int rowBase = m0 + (wv << 4) + ((lane & 16) ? 8 : 0);
  int colB = n0 + (lane & 15);
#pragma unroll
  for (int r = 0; r < 8; ++r) {
    int row = rowBase + r;
    if (!full && row >= M) continue;
#pragma unroll
    for (int t = 0; t < 4; ++t) {
      int col = colB + 16 * t;
      float v = acc[t].f[r];
      if (FLAGS & 1) v += bias[col];
      if (FLAGS & 2) v = silu_f(v);
      size_t o = (size_t)row * H + col;
      if (FLAGS & 4) v *= mulp[o];
      if (FLAGS & 8) v += addp[o];
      C[o] = v;
    }
  }
}

// ---------------------------------------------------------------------------
// f32-B GEMM kept only for the tiny K=6 radial layers (single K-step).
// ---------------------------------------------------------------------------
template <int FLAGS>
__global__ __launch_bounds__(256) void gemm_kernel(
    const float* __restrict__ A, const float* __restrict__ B,
    const float* __restrict__ bias, const float* __restrict__ mulp,
    const float* __restrict__ addp, float* __restrict__ C,
    int M, int N, int K) {
  __shared__ _Float16 As[128][LDSTRIDE];
  __shared__ _Float16 Bt[64][LDSTRIDE];
  int tid = threadIdx.x, lane = tid & 31, wv = tid >> 5;
  int m0 = blockIdx.x * 128, n0 = blockIdx.y * 64;
  F8 acc[4];
#pragma unroll
  for (int t = 0; t < 4; ++t)
#pragma unroll
    for (int r = 0; r < 8; ++r) acc[t].f[r] = 0.f;

  for (int k0 = 0; k0 < K; k0 += 32) {
    __syncthreads();
    for (int idx = tid; idx < 128 * 32; idx += 256) {
      int r = idx >> 5, cc = idx & 31;
      int gr = m0 + r, gc = k0 + cc;
      As[r][cc] = (gr < M && gc < K) ? (_Float16)A[(size_t)gr * K + gc] : (_Float16)0.f;
    }
    for (int idx = tid; idx < 64 * 16; idx += 256) {
      int n = idx & 63, k2 = (idx >> 6) << 1;
      int gk = k0 + k2, gn = n0 + n;
      float f0 = (gk < K) ? B[(size_t)gk * N + gn] : 0.f;
      float f1 = (gk + 1 < K) ? B[(size_t)(gk + 1) * N + gn] : 0.f;
      *(v2h*)&Bt[n][k2] = pk2(f0, f1);
    }
    __syncthreads();

    H16 a, b[4];
    int am = (wv << 4) + (lane & 15);
    int koffA = (lane & 16) ? 8 : 0;
    a.q[0] = *(const v8h*)&As[am][koffA];
    a.q[1] = *(const v8h*)&As[am][koffA + 16];
    int bn = lane & 15, koffB = (lane & 16) ? 8 : 0;
#pragma unroll
    for (int t = 0; t < 4; ++t) {
      b[t].q[0] = *(const v8h*)&Bt[bn + 16 * t][koffB];
      b[t].q[1] = *(const v8h*)&Bt[bn + 16 * t][koffB + 16];
    }
#pragma unroll
    for (int t = 0; t < 4; ++t)
      acc[t].v = __builtin_amdgcn_wmma_f32_16x16x32_f16(false, a.v, false, b[t].v,
                                                        (short)0, acc[t].v, false, false);
  }

  bool full = (m0 + 128 <= M);
  int rowBase = m0 + (wv << 4) + ((lane & 16) ? 8 : 0);
#pragma unroll
  for (int r = 0; r < 8; ++r) {
    int row = rowBase + r;
    if (!full && row >= M) continue;
#pragma unroll
    for (int t = 0; t < 4; ++t) {
      int col = n0 + (lane & 15) + 16 * t;
      float v = acc[t].f[r];
      if (FLAGS & 1) v += bias[col];
      if (FLAGS & 2) v = silu_f(v);
      size_t o = (size_t)row * N + col;
      if (FLAGS & 4) v *= mulp[o];
      if (FLAGS & 8) v += addp[o];
      C[o] = v;
    }
  }
}

// ---------------------------------------------------------------------------
// Bilinear einsum as fused-gather WMMA GEMM (f16 W_bil, async B staging,
// LDS double buffer):
//   A[w, j*128+l] = sbf_p[w,j] * x_kj[idx_kj[w], l]
//   out[idx_ji[w], i] += ...   (atomic scatter == segment_sum over idx_ji)
// ---------------------------------------------------------------------------
__global__ __launch_bounds__(256) void bilin_kernel(
    const float* __restrict__ sbfp, const float* __restrict__ xkj,
    const int* __restrict__ ikj, const int* __restrict__ iji,
    const _Float16* __restrict__ Wbt, float* __restrict__ outE, int T) {
  __shared__ _Float16 As[2][128][LDSTRIDE];
  __shared__ _Float16 Bs[2][64][LDSTRIDE];
  int tid = threadIdx.x, lane = tid & 31, wv = tid >> 5;
  int m0 = blockIdx.x * 128, n0 = blockIdx.y * 64;
  bool full = (m0 + 128 <= T);
  F8 acc[4];
#pragma unroll
  for (int t = 0; t < 4; ++t)
#pragma unroll
    for (int r = 0; r < 8; ++r) acc[t].f[r] = 0.f;

  int bn_row = tid >> 2, bn_ch = (tid & 3) << 3;
  int ar = tid >> 3, ac4 = (tid & 7) << 2;

  auto stage = [&](int k0, int buf) {
    async_b128(&Bs[buf][bn_row][bn_ch], Wbt,
               (unsigned)(((size_t)(n0 + bn_row) * (NB * H) + k0 + bn_ch) * sizeof(_Float16)));
    int jg = k0 >> 7, l0 = k0 & 127;
    if (full) {
#pragma unroll
      for (int it = 0; it < 4; ++it) {
        int r = ar + it * 32;
        int w = m0 + r;
        int kj = ikj[w];
        float sv = sbfp[(size_t)w * NB + jg];
        const float4 f = *(const float4*)&xkj[(size_t)kj * H + l0 + ac4];
        V4P u;
        u.v2[0] = pk2(sv * f.x, sv * f.y);
        u.v2[1] = pk2(sv * f.z, sv * f.w);
        *(v4h*)&As[buf][r][ac4] = u.v4;
      }
    } else {
#pragma unroll
      for (int it = 0; it < 4; ++it) {
        int r = ar + it * 32;
        int w = m0 + r;
        V4P u;
        if (w < T) {
          int kj = ikj[w];
          float sv = sbfp[(size_t)w * NB + jg];
          const float4 f = *(const float4*)&xkj[(size_t)kj * H + l0 + ac4];
          u.v2[0] = pk2(sv * f.x, sv * f.y);
          u.v2[1] = pk2(sv * f.z, sv * f.w);
        } else {
          u.v2[0] = pk2(0.f, 0.f);
          u.v2[1] = pk2(0.f, 0.f);
        }
        *(v4h*)&As[buf][r][ac4] = u.v4;
      }
    }
  };

  stage(0, 0);
  int cur = 0;
  for (int k0 = 0; k0 < NB * H; k0 += 32, cur ^= 1) {
    wait_async0();
    __syncthreads();
    if (k0 + 32 < NB * H) stage(k0 + 32, cur ^ 1);

    H16 a, b[4];
    int am = (wv << 4) + (lane & 15);
    int koffA = (lane & 16) ? 8 : 0;
    a.q[0] = *(const v8h*)&As[cur][am][koffA];
    a.q[1] = *(const v8h*)&As[cur][am][koffA + 16];
    int bn = lane & 15, koffB = (lane & 16) ? 8 : 0;
#pragma unroll
    for (int t = 0; t < 4; ++t) {
      b[t].q[0] = *(const v8h*)&Bs[cur][bn + 16 * t][koffB];
      b[t].q[1] = *(const v8h*)&Bs[cur][bn + 16 * t][koffB + 16];
    }
#pragma unroll
    for (int t = 0; t < 4; ++t)
      acc[t].v = __builtin_amdgcn_wmma_f32_16x16x32_f16(false, a.v, false, b[t].v,
                                                        (short)0, acc[t].v, false, false);
  }

  int rowBase = m0 + (wv << 4) + ((lane & 16) ? 8 : 0);
  int colB = n0 + (lane & 15);
#pragma unroll
  for (int r = 0; r < 8; ++r) {
    int w = rowBase + r;
    if (!full && w >= T) continue;
    int e = iji[w];
#pragma unroll
    for (int t = 0; t < 4; ++t)
      atomicAdd(&outE[(size_t)e * H + colB + 16 * t], acc[t].f[r]);
  }
}

// ---------------------------------------------------------------------------
// Host orchestration
// ---------------------------------------------------------------------------
enum { FB = 1, FS = 2, FM = 4, FA = 8 };

static inline void launch_gemmh(hipStream_t s, const float* A, const _Float16* Wt,
                                const float* bias, const float* mulp, const float* addp,
                                float* C, int M, int K, int flags) {
  dim3 grid((M + 127) / 128, 2), blk(256);
  switch (flags) {
    case FB|FS:    gemmh_kernel<FB|FS><<<grid, blk, 0, s>>>(A, Wt, bias, mulp, addp, C, M, K); break;
    case FB|FS|FM: gemmh_kernel<FB|FS|FM><<<grid, blk, 0, s>>>(A, Wt, bias, mulp, addp, C, M, K); break;
    case FB|FS|FA: gemmh_kernel<FB|FS|FA><<<grid, blk, 0, s>>>(A, Wt, bias, mulp, addp, C, M, K); break;
    default:       gemmh_kernel<0><<<grid, blk, 0, s>>>(A, Wt, bias, mulp, addp, C, M, K); break;
  }
}

static inline void launch_gemm(hipStream_t s, const float* A, const float* B,
                               const float* bias, const float* mulp, const float* addp,
                               float* C, int M, int N, int K, int flags) {
  dim3 grid((M + 127) / 128, N / 64), blk(256);
  switch (flags) {
    case 0:     gemm_kernel<0><<<grid, blk, 0, s>>>(A, B, bias, mulp, addp, C, M, N, K); break;
    case FM:    gemm_kernel<FM><<<grid, blk, 0, s>>>(A, B, bias, mulp, addp, C, M, N, K); break;
    case FB|FS: gemm_kernel<FB|FS><<<grid, blk, 0, s>>>(A, B, bias, mulp, addp, C, M, N, K); break;
    default:    gemm_kernel<0><<<grid, blk, 0, s>>>(A, B, bias, mulp, addp, C, M, N, K); break;
  }
}

struct IBP {
  const float *Wbil, *jiW, *jib, *kjW, *kjb, *linW, *linb, *rbfW;
  const float *rasW[2][2], *rasB[2][2], *rbsW[2], *rbsB[2], *sbfW;
};
struct OBP {
  const float* finalW;
  const float *lW[3], *lB[3];
  const float* rbfW;
};

extern "C" void kernel_launch(void* const* d_in, const int* in_sizes, int n_in,
                              void* d_out, int out_size, void* d_ws, size_t ws_size,
                              hipStream_t stream) {
  // --- input parsing (jax pytree flatten: dicts in sorted-key order) ---
  int c = 0;
  auto Fp = [&]() { return (const float*)d_in[c++]; };
  const float* pos = Fp();                                   // "pos"
  const float* emb = Fp();                                   // params["emb"]
  const float* emblinW = Fp(); const float* emblinb = Fp();  // params["emb_lin"]
  const float* embrbfW = Fp(); const float* embrbfb = Fp();  // params["emb_rbf"]
  IBP ib[6];
  for (int b = 0; b < 6; ++b) {                              // params["int"][b]
    ib[b].Wbil = Fp();
    ib[b].jiW = Fp(); ib[b].jib = Fp();
    ib[b].kjW = Fp(); ib[b].kjb = Fp();
    ib[b].linW = Fp(); ib[b].linb = Fp();
    ib[b].rbfW = Fp();
    for (int r = 0; r < 2; ++r)
      for (int l = 0; l < 2; ++l) { ib[b].rasW[r][l] = Fp(); ib[b].rasB[r][l] = Fp(); }
    for (int l = 0; l < 2; ++l) { ib[b].rbsW[l] = Fp(); ib[b].rbsB[l] = Fp(); }
    ib[b].sbfW = Fp();
  }
  OBP ob[7];
  for (int b = 0; b < 7; ++b) {                              // params["out"][b]
    ob[b].finalW = Fp();
    for (int l = 0; l < 3; ++l) { ob[b].lW[l] = Fp(); ob[b].lB[l] = Fp(); }
    ob[b].rbfW = Fp();
  }
  const float* rbf_freq = Fp();                              // params["rbf_freq"]
  const int* z    = (const int*)d_in[c]; int i_z = c; ++c;
  const int* esrc = (const int*)d_in[c]; int i_e = c; ++c;
  const int* edst = (const int*)d_in[c]; ++c;
  const int* ikj  = (const int*)d_in[c]; int i_t = c; ++c;
  const int* iji  = (const int*)d_in[c]; ++c;
  const int* batch = (const int*)d_in[c]; ++c;
  (void)n_in;

  const int Nn = in_sizes[i_z];
  const int E  = in_sizes[i_e];
  const int T  = in_sizes[i_t];

  // --- workspace carve-out (256B-aligned) ---
  char* wp = (char*)d_ws;
  auto alloc = [&](size_t nf) {
    float* p = (float*)wp;
    wp += ((nf * sizeof(float) + 255) & ~(size_t)255);
    return p;
  };
  auto allocH = [&](size_t nh) {
    _Float16* p = (_Float16*)wp;
    wp += ((nh * sizeof(_Float16) + 255) & ~(size_t)255);
    return p;
  };
  float* norm42 = alloc(42);
  float* rbf  = alloc((size_t)E * NR);
  float* rad  = alloc((size_t)E * 42);
  float* cbfB = alloc((size_t)T * NS);
  float* sbfp = alloc((size_t)T * NB);
  float* rbfe = alloc((size_t)E * H);
  float* cat  = alloc((size_t)E * 3 * H);
  float* xji  = alloc((size_t)E * H);
  float* xkj  = alloc((size_t)E * H);
  float* rbfp = alloc((size_t)E * H);
  float* t0   = alloc((size_t)E * H);
  float* t1   = alloc((size_t)E * H);
  float* t2   = alloc((size_t)E * H);
  float* gbuf = alloc((size_t)E * H);
  float* tn0  = alloc((size_t)Nn * H);
  float* tn1  = alloc((size_t)Nn * H);
  // f16 weight pool
  _Float16* emblinWt = allocH((size_t)3 * H * H);
  _Float16 *jiWt[6], *kjWt[6], *linWt[6], *rbsWt[6][2], *rasWt[6][2][2], *WbilH[6];
  for (int b = 0; b < 6; ++b) {
    jiWt[b]  = allocH((size_t)H * H);
    kjWt[b]  = allocH((size_t)H * H);
    linWt[b] = allocH((size_t)H * H);
    for (int l = 0; l < 2; ++l) rbsWt[b][l] = allocH((size_t)H * H);
    for (int r = 0; r < 2; ++r)
      for (int l = 0; l < 2; ++l) rasWt[b][r][l] = allocH((size_t)H * H);
    WbilH[b] = allocH((size_t)H * NB * H);
  }
  _Float16* obWt[7][3];
  for (int b = 0; b < 7; ++b)
    for (int l = 0; l < 3; ++l) obWt[b][l] = allocH((size_t)H * H);
  (void)ws_size;

  float* P = (float*)d_out;
  hipMemsetAsync(P, 0, (size_t)out_size * sizeof(float), stream);

  // --- batched weight conversions (f32 -> transposed f16) ---
  ConvTab tab;
  int nw = 0;
  auto addW = [&](const float* W, _Float16* Wt) { tab.src[nw] = W; tab.dst[nw] = Wt; ++nw; };
  for (int b = 0; b < 6; ++b) {
    addW(ib[b].jiW, jiWt[b]);
    addW(ib[b].kjW, kjWt[b]);
    addW(ib[b].linW, linWt[b]);
    for (int l = 0; l < 2; ++l) addW(ib[b].rbsW[l], rbsWt[b][l]);
    for (int r = 0; r < 2; ++r)
      for (int l = 0; l < 2; ++l) addW(ib[b].rasW[r][l], rasWt[b][r][l]);
  }
  for (int b = 0; b < 7; ++b)
    for (int l = 0; l < 3; ++l) addW(ob[b].lW[l], obWt[b][l]);
  for (int i = nw; i < MAXW; ++i) { tab.src[i] = nullptr; tab.dst[i] = nullptr; }
  convT128_kernel<<<dim3((H * 64) / 256, nw), 256, 0, stream>>>(tab);
  {
    ConvFTab ftab;
    for (int b = 0; b < 6; ++b) { ftab.src[b] = ib[b].Wbil; ftab.dst[b] = WbilH[b]; }
    convF_batch_kernel<<<dim3((H * NB * H / 2) / 256, 6), 256, 0, stream>>>(ftab);
  }
  {
    int tot = H * ((3 * H) >> 1);
    convT_kernel<<<(tot + 255) / 256, 256, 0, stream>>>(emblinW, emblinWt, 3 * H, H);
  }

  // --- geometry ---
  norm_kernel<<<1, 64, 0, stream>>>(norm42);
  edge_kernel<<<(E + 255) / 256, 256, 0, stream>>>(pos, esrc, edst, rbf_freq, norm42, rbf, rad, E);
  trip_kernel<<<(T + 255) / 256, 256, 0, stream>>>(pos, esrc, edst, ikj, iji, cbfB, T);

  // --- embedding block ---
  launch_gemm(stream, rbf, embrbfW, embrbfb, nullptr, nullptr, rbfe, E, H, NR, FB | FS);
  concat_kernel<<<(int)(((size_t)E * H + 255) / 256), 256, 0, stream>>>(emb, z, esrc, edst, rbfe, cat, E);
  float* x = t0;
  launch_gemmh(stream, cat, emblinWt, emblinb, nullptr, nullptr, x, E, 3 * H, FB | FS);

  auto out_block = [&](const OBP& o, int bi, const float* xcur) {
    launch_gemm(stream, rbf, o.rbfW, nullptr, xcur, nullptr, gbuf, E, H, NR, FM);
    hipMemsetAsync(tn0, 0, (size_t)Nn * H * sizeof(float), stream);
    segsum_kernel<<<(int)(((size_t)E * H + 255) / 256), 256, 0, stream>>>(gbuf, edst, tn0, E);
    launch_gemmh(stream, tn0, obWt[bi][0], o.lB[0], nullptr, nullptr, tn1, Nn, H, FB | FS);
    launch_gemmh(stream, tn1, obWt[bi][1], o.lB[1], nullptr, nullptr, tn0, Nn, H, FB | FS);
    launch_gemmh(stream, tn0, obWt[bi][2], o.lB[2], nullptr, nullptr, tn1, Nn, H, FB | FS);
    final_kernel<<<(Nn + 255) / 256, 256, 0, stream>>>(tn1, o.finalW, batch, P, Nn);
  };

  out_block(ob[0], 0, x);

  float* f0 = t1;
  float* f1 = t2;
  for (int b = 0; b < 6; ++b) {
    const IBP& p = ib[b];
    launch_gemm(stream, rbf, p.rbfW, nullptr, nullptr, nullptr, rbfp, E, H, NR, 0);
    launch_gemmh(stream, x, jiWt[b], p.jib, nullptr, nullptr, xji, E, H, FB | FS);
    launch_gemmh(stream, x, kjWt[b], p.kjb, rbfp, nullptr, xkj, E, H, FB | FS | FM);
    sbfp_kernel<<<(T + 255) / 256, 256, 0, stream>>>(rad, cbfB, ikj, p.sbfW, sbfp, T);

    float* t = f0;  // t = x_ji + segment_sum(m)
    float* u = f1;
    hipMemcpyAsync(t, xji, (size_t)E * H * sizeof(float), hipMemcpyDeviceToDevice, stream);
    bilin_kernel<<<dim3((T + 127) / 128, 2), 256, 0, stream>>>(sbfp, xkj, ikj, iji, WbilH[b], t, T);

    // res_bs (1): t = t + silu(silu(t@W0+b0)@W1+b1)
    launch_gemmh(stream, t, rbsWt[b][0], p.rbsB[0], nullptr, nullptr, u, E, H, FB | FS);
    launch_gemmh(stream, u, rbsWt[b][1], p.rbsB[1], nullptr, t, t, E, H, FB | FS | FA);
    // lin: t = silu(t@W+b) + x
    launch_gemmh(stream, t, linWt[b], p.linb, nullptr, x, u, E, H, FB | FS | FA);
    float* xold = x;
    float* tc = u;
    // res_as (2)
    launch_gemmh(stream, tc, rasWt[b][0][0], p.rasB[0][0], nullptr, nullptr, xold, E, H, FB | FS);
    launch_gemmh(stream, xold, rasWt[b][0][1], p.rasB[0][1], nullptr, tc, tc, E, H, FB | FS | FA);
    launch_gemmh(stream, tc, rasWt[b][1][0], p.rasB[1][0], nullptr, nullptr, t, E, H, FB | FS);
    launch_gemmh(stream, t, rasWt[b][1][1], p.rasB[1][1], nullptr, tc, tc, E, H, FB | FS | FA);

    x = tc;      // new x
    f0 = xold;   // freed
    f1 = t;      // freed

    out_block(ob[b + 1], b + 1, x);
  }
}
